// Attention_3788161155740
// MI455X (gfx1250) — matrix operand order repaired and checked by reading, not compiled, not run on hardware
//
#include <hip/hip_runtime.h>
#include <hip/hip_bf16.h>
#include <math.h>

// ---------------------------------------------------------------------------
// MHA for B=8, W=64, S=256, E=512, H=8, D=64 on gfx1250 (wave32, WMMA, TDM).
// ---------------------------------------------------------------------------

typedef __attribute__((ext_vector_type(16))) _Float16     v16h;
typedef __attribute__((ext_vector_type(8)))  float        v8f;
typedef __attribute__((ext_vector_type(4)))  float        f32x4;
typedef __attribute__((ext_vector_type(4)))  unsigned int u32x4;
typedef __attribute__((ext_vector_type(8)))  int          i32x8;
typedef __attribute__((ext_vector_type(4)))  int          i32x4;

union Frag {
    v16h  h;
    f32x4 q[2];
};

#define NTOK   131072      // B*W*S
#define EDIM   512
#define SDIM   256
#define HEADS  8
#define DHEAD  64
#define ATT_SCALE 0.125f   // 64^-0.5

static __device__ __forceinline__ v8f wmma_f16(const Frag& a, const Frag& b, v8f c) {
    return __builtin_amdgcn_wmma_f32_16x16x32_f16(false, a.h, false, b.h,
                                                  (short)0, c, false, false);
}

// LDS byte offset of a __shared__ object (generic addr low 32 bits = LDS offset)
static __device__ __forceinline__ unsigned lds_off(const void* p) {
    return (unsigned)(uintptr_t)p;
}

// Async copy 16 bytes global -> LDS per lane (ASYNCcnt-tracked).
static __device__ __forceinline__ void async_b128(unsigned lds, const void* g) {
    asm volatile("global_load_async_to_lds_b128 %0, %1, off"
                 :: "v"(lds), "v"((unsigned long long)(uintptr_t)g)
                 : "memory");
}
static __device__ __forceinline__ void wait_async0() {
    asm volatile("s_wait_asynccnt 0x0" ::: "memory");
}

// ---------------------------------------------------------------------------
// f32 -> f16 elementwise pre-convert (x and the 4 weight matrices)
// ---------------------------------------------------------------------------
__global__ __launch_bounds__(256)
void cvt_f32_f16(const float* __restrict__ s, _Float16* __restrict__ d, int n8) {
    int i = blockIdx.x * 256 + threadIdx.x;
    if (i >= n8) return;
    const f32x4* sp = (const f32x4*)s + 2 * (size_t)i;
    f32x4 a = sp[0], b = sp[1];
    union { _Float16 h[8]; f32x4 v; } u;
    u.h[0] = (_Float16)a[0]; u.h[1] = (_Float16)a[1];
    u.h[2] = (_Float16)a[2]; u.h[3] = (_Float16)a[3];
    u.h[4] = (_Float16)b[0]; u.h[5] = (_Float16)b[1];
    u.h[6] = (_Float16)b[2]; u.h[7] = (_Float16)b[3];
    ((f32x4*)d)[i] = u.v;
}

// ---------------------------------------------------------------------------
// GEMM:  out[m][n] = (sum_k A16[m][k] * W16[n][k] + bias[n]) * scale
// M = NTOK, N = K = 512.  Block = 128 threads (4 waves), tile 128x128.
// A/B tiles staged via async global->LDS copies (no VALU in the copy path).
// ---------------------------------------------------------------------------
template <bool OUT_F16>
__global__ __launch_bounds__(128)
void gemm512_kernel(const _Float16* __restrict__ A16, const _Float16* __restrict__ W16,
                    const float* __restrict__ bias, void* __restrict__ out,
                    float scale)
{
    __shared__ _Float16 As[128][40];   // 128 rows x 32 k halves, padded
    __shared__ _Float16 Bs[128][40];   // 128 n-rows x 32 k halves, padded

    const int tid   = threadIdx.x;
    const int mBase = blockIdx.x * 128;
    const int nBase = blockIdx.y * 128;
    const int wid   = tid >> 5;
    const int lane  = tid & 31;
    const int ln    = lane & 15;
    const int lh    = lane >> 4;
    const int wrow  = (wid >> 1) * 64;
    const int wcol  = (wid & 1) * 64;

    v8f acc[4][4];
#pragma unroll
    for (int i = 0; i < 4; ++i)
#pragma unroll
        for (int j = 0; j < 4; ++j) acc[i][j] = (v8f)0.0f;

    for (int k0 = 0; k0 < EDIM; k0 += 32) {
        __syncthreads();
        {
            const _Float16* srcA = A16 + (size_t)(mBase + tid) * EDIM + k0;
            const unsigned la = lds_off(&As[tid][0]);
#pragma unroll
            for (int i = 0; i < 4; ++i) async_b128(la + i * 16, srcA + i * 8);
            const _Float16* srcB = W16 + (size_t)(nBase + tid) * EDIM + k0;
            const unsigned lb = lds_off(&Bs[tid][0]);
#pragma unroll
            for (int i = 0; i < 4; ++i) async_b128(lb + i * 16, srcB + i * 8);
        }
        wait_async0();
        __syncthreads();

        Frag fa[4], fb[4];
#pragma unroll
        for (int i = 0; i < 4; ++i) {
            const _Float16* p = &As[wrow + i * 16 + ln][lh * 8];
            fa[i].q[0] = *(const f32x4*)p;
            fa[i].q[1] = *(const f32x4*)(p + 16);
        }
#pragma unroll
        for (int j = 0; j < 4; ++j) {
            const _Float16* p = &Bs[wcol + j * 16 + ln][lh * 8];
            fb[j].q[0] = *(const f32x4*)p;
            fb[j].q[1] = *(const f32x4*)(p + 16);
        }
#pragma unroll
        for (int i = 0; i < 4; ++i)
#pragma unroll
            for (int j = 0; j < 4; ++j)
                acc[i][j] = wmma_f16(fa[i], fb[j], acc[i][j]);
    }

    // ---- epilogue: bias + scale, store ----
#pragma unroll
    for (int j = 0; j < 4; ++j) {
        const int col = nBase + wcol + j * 16 + ln;
        const float bv = bias[col];
#pragma unroll
        for (int i = 0; i < 4; ++i) {
            const int row0 = mBase + wrow + i * 16 + 8 * lh;
#pragma unroll
            for (int r = 0; r < 8; ++r) {
                float v = (acc[i][j][r] + bv) * scale;
                size_t idx = (size_t)(row0 + r) * EDIM + col;
                if (OUT_F16) ((_Float16*)out)[idx] = (_Float16)v;
                else         ((float*)out)[idx]    = v;
            }
        }
    }
}

// ---------------------------------------------------------------------------
// Flash attention per (b, w, h, half-of-S). Block = 256 threads (8 waves).
// K tile staged by the Tensor Data Mover (with LDS row padding); V tile
// transposed manually; P round-trips per-wave LDS for C->A layout change.
// ---------------------------------------------------------------------------
__global__ __launch_bounds__(256)
void attn_kernel(const _Float16* __restrict__ q16, const _Float16* __restrict__ k16,
                 const _Float16* __restrict__ v16, const float* __restrict__ mask,
                 const float* __restrict__ pos_bias, _Float16* __restrict__ z16)
{
    __shared__ _Float16 Ks[128][72];       // k rows (j-major), 128B data + 16B pad
    __shared__ _Float16 Vt[64][136];       // v transposed: [d][j], padded
    __shared__ _Float16 Ps[8][16][40];     // per-wave P tile (16 x 32), padded

    const int bid   = blockIdx.x;
    const int halfb = bid & 1;
    const int h     = (bid >> 1) & 7;
    const int w     = (bid >> 4) & 63;
    const int b     = bid >> 10;
    const size_t tokBase = ((size_t)b * 64 + w) * SDIM;

    const int tid  = threadIdx.x;
    const int wid  = tid >> 5;
    const int lane = tid & 31;
    const int ln   = lane & 15;
    const int lh   = lane >> 4;
    const int i0   = halfb * 128 + wid * 16;   // query-row base for this wave

    // ---- q fragments (pre-scaled by ATT_SCALE in the projection) ----
    Frag qa[2];
    {
        const _Float16* qrow = q16 + (tokBase + i0 + ln) * EDIM + h * DHEAD;
#pragma unroll
        for (int kk = 0; kk < 2; ++kk) {
            const _Float16* p = qrow + kk * 32 + lh * 8;
            qa[kk].q[0] = *(const f32x4*)p;
            qa[kk].q[1] = *(const f32x4*)(p + 16);
        }
    }

    float row_m[8], row_l[8];
#pragma unroll
    for (int r = 0; r < 8; ++r) { row_m[r] = -INFINITY; row_l[r] = 0.0f; }
    v8f zacc[4];
#pragma unroll
    for (int t = 0; t < 4; ++t) zacc[t] = (v8f)0.0f;

    const float* pbBase = pos_bias + ((size_t)h * SDIM + (i0 + 8 * lh)) * SDIM;
    const float* mkBase = mask     + ((size_t)w * SDIM + (i0 + 8 * lh)) * SDIM;

    for (int jc = 0; jc < SDIM; jc += 128) {
        // prefetch bias/mask rows for this chunk
#pragma unroll
        for (int r = 0; r < 8; ++r) {
            __builtin_prefetch(&pbBase[(size_t)r * SDIM + jc], 0, 0);
            __builtin_prefetch(&mkBase[(size_t)r * SDIM + jc], 0, 0);
        }
        __syncthreads();

        // ---- stage K chunk: 128 rows x 64 halves, row-padded to 72 ----
#if __has_builtin(__builtin_amdgcn_tensor_load_to_lds)
        if (wid == 0) {
            const unsigned long long ga =
                (unsigned long long)(uintptr_t)(k16 + (tokBase + jc) * EDIM + h * DHEAD);
            u32x4 g0;
            g0[0] = 1u;                                   // count=1 valid descriptor
            g0[1] = lds_off(&Ks[0][0]);                   // lds_addr
            g0[2] = (unsigned)ga;                         // global_addr[31:0]
            g0[3] = (unsigned)((ga >> 32) & 0x1FFFFFFu) | (2u << 30);  // addr hi | type=2
            i32x8 g1;
            g1[0] = (int)((1u << 16)      // data_size = 2 bytes
                        | (1u << 20)      // pad_enable
                        | (4u << 22)      // pad_interval: 32 DWORDs = 128B
                        | (3u << 25));    // pad_amount:   4 DWORDs = 16B
            g1[1] = (int)(64u << 16);     // tensor_dim0 = 64 (lo16)
            g1[2] = (int)(128u << 16);    // tensor_dim1 = 128 (lo16)
            g1[3] = (int)(64u << 16);     // tile_dim0 = 64
            g1[4] = (int)128u;            // tile_dim1 = 128
            g1[5] = (int)512u;            // tensor_dim0_stride = 512 elements
            g1[6] = 0;
            g1[7] = 0;
            i32x4 g2 = (i32x4)0;
            i32x4 g3 = (i32x4)0;
            i32x8 g4 = (i32x8)0;          // 6-arg toolchain form: extra group
            __builtin_amdgcn_tensor_load_to_lds(g0, g1, g2, g3, g4, 0);
            __builtin_amdgcn_s_wait_tensorcnt(0);
        }
#else
        {
            const int r = tid >> 1, part = tid & 1;
            const _Float16* src = k16 + (tokBase + jc + r) * EDIM + h * DHEAD + part * 32;
            const unsigned lk = lds_off(&Ks[r][part * 32]);
#pragma unroll
            for (int i = 0; i < 4; ++i) async_b128(lk + i * 16, src + i * 8);
        }
#endif
        // ---- stage V chunk transposed: Vt[d][j] ----
        {
            const int r = tid >> 1, part = tid & 1;
            const _Float16* src = v16 + (tokBase + jc + r) * EDIM + h * DHEAD + part * 32;
            _Float16 tmp[32];
            const f32x4* s4 = (const f32x4*)src;
            f32x4* t4 = (f32x4*)tmp;
#pragma unroll
            for (int i = 0; i < 4; ++i) t4[i] = s4[i];
#pragma unroll
            for (int d = 0; d < 32; ++d) Vt[part * 32 + d][r] = tmp[d];
        }
        wait_async0();
        __syncthreads();

        for (int jp = 0; jp < 4; ++jp) {        // 32 j-columns per pass
#pragma unroll
            for (int s = 0; s < 2; ++s) {       // two 16-wide score tiles
                const int jt = jp * 32 + s * 16;
                Frag kb[2];
#pragma unroll
                for (int kk = 0; kk < 2; ++kk) {
                    const _Float16* p = &Ks[jt + ln][kk * 32 + lh * 8];
                    kb[kk].q[0] = *(const f32x4*)p;
                    kb[kk].q[1] = *(const f32x4*)(p + 16);
                }
                v8f sc = (v8f)0.0f;
                sc = wmma_f16(qa[0], kb[0], sc);
                sc = wmma_f16(qa[1], kb[1], sc);

                const int jg = jc + jt + ln;    // global column for this lane
                float sv[8];
#pragma unroll
                for (int r = 0; r < 8; ++r)
                    sv[r] = sc[r] + pbBase[(size_t)r * SDIM + jg]
                                  + mkBase[(size_t)r * SDIM + jg];

#pragma unroll
                for (int r = 0; r < 8; ++r) {
                    float mx = sv[r];
#pragma unroll
                    for (int off = 1; off < 16; off <<= 1)
                        mx = fmaxf(mx, __shfl_xor(mx, off, 16));
                    const float mnew = fmaxf(row_m[r], mx);
                    const float rs   = __expf(row_m[r] - mnew);
                    const float pr   = __expf(sv[r] - mnew);
                    float sum = pr;
#pragma unroll
                    for (int off = 1; off < 16; off <<= 1)
                        sum += __shfl_xor(sum, off, 16);
                    row_l[r] = row_l[r] * rs + sum;
                    row_m[r] = mnew;
#pragma unroll
                    for (int t = 0; t < 4; ++t) zacc[t][r] *= rs;
                    Ps[wid][r + 8 * lh][s * 16 + ln] = (_Float16)pr;
                }
            }
            // make per-wave P tile visible across lanes of this wave
            __builtin_amdgcn_fence(__ATOMIC_ACQ_REL, "wavefront");

            Frag pa;
            {
                const _Float16* pp = &Ps[wid][ln][lh * 8];
                pa.q[0] = *(const f32x4*)pp;
                pa.q[1] = *(const f32x4*)(pp + 16);
            }
#pragma unroll
            for (int t = 0; t < 4; ++t) {
                Frag vb;
                const _Float16* vp = &Vt[t * 16 + ln][jp * 32 + lh * 8];
                vb.q[0] = *(const f32x4*)vp;
                vb.q[1] = *(const f32x4*)(vp + 16);
                zacc[t] = wmma_f16(pa, vb, zacc[t]);
            }
            __builtin_amdgcn_fence(__ATOMIC_ACQ_REL, "wavefront");
        }
    }

    // ---- normalize and store z (f16) ----
    float inv_l[8];
#pragma unroll
    for (int r = 0; r < 8; ++r) inv_l[r] = 1.0f / row_l[r];
#pragma unroll
    for (int t = 0; t < 4; ++t) {
#pragma unroll
        for (int r = 0; r < 8; ++r) {
            size_t idx = (tokBase + i0 + r + 8 * lh) * (size_t)EDIM
                       + h * DHEAD + t * 16 + ln;
            z16[idx] = (_Float16)(zacc[t][r] * inv_l[r]);
        }
    }
}

// ---------------------------------------------------------------------------
// Launch
// ---------------------------------------------------------------------------
extern "C" void kernel_launch(void* const* d_in, const int* in_sizes, int n_in,
                              void* d_out, int out_size, void* d_ws, size_t ws_size,
                              hipStream_t stream) {
    (void)in_sizes; (void)n_in; (void)out_size; (void)ws_size;

    const float* x        = (const float*)d_in[0];
    const float* mask     = (const float*)d_in[1];
    const float* Wq       = (const float*)d_in[2];
    const float* bq       = (const float*)d_in[3];
    const float* Wk       = (const float*)d_in[4];
    const float* bk       = (const float*)d_in[5];
    const float* Wv       = (const float*)d_in[6];
    const float* bv       = (const float*)d_in[7];
    const float* Wp       = (const float*)d_in[8];
    const float* bp       = (const float*)d_in[9];
    const float* pos_bias = (const float*)d_in[10];
    float* out = (float*)d_out;

    // workspace layout (f16 buffers)
    const size_t tokBytes = (size_t)NTOK * EDIM * sizeof(_Float16); // 128 MiB
    const size_t wBytes   = (size_t)EDIM * EDIM * sizeof(_Float16); // 512 KiB
    char* p = (char*)d_ws;
    _Float16* x16  = (_Float16*)p; p += tokBytes;
    _Float16* q16  = (_Float16*)p; p += tokBytes;
    _Float16* k16  = (_Float16*)p; p += tokBytes;
    _Float16* v16  = (_Float16*)p; p += tokBytes;
    _Float16* z16  = (_Float16*)p; p += tokBytes;
    _Float16* wq16 = (_Float16*)p; p += wBytes;
    _Float16* wk16 = (_Float16*)p; p += wBytes;
    _Float16* wv16 = (_Float16*)p; p += wBytes;
    _Float16* wp16 = (_Float16*)p; p += wBytes;

    // pre-convert inputs to f16
    const int x8 = NTOK * EDIM / 8;           // 8,388,608
    const int w8 = EDIM * EDIM / 8;           // 32,768
    cvt_f32_f16<<<(x8 + 255) / 256, 256, 0, stream>>>(x,  x16,  x8);
    cvt_f32_f16<<<(w8 + 255) / 256, 256, 0, stream>>>(Wq, wq16, w8);
    cvt_f32_f16<<<(w8 + 255) / 256, 256, 0, stream>>>(Wk, wk16, w8);
    cvt_f32_f16<<<(w8 + 255) / 256, 256, 0, stream>>>(Wv, wv16, w8);
    cvt_f32_f16<<<(w8 + 255) / 256, 256, 0, stream>>>(Wp, wp16, w8);

    dim3 gGemm(NTOK / 128, EDIM / 128);
    // q/k/v projections (q pre-scaled by D^-0.5)
    gemm512_kernel<true><<<gGemm, 128, 0, stream>>>(x16, wq16, bq, (void*)q16, ATT_SCALE);
    gemm512_kernel<true><<<gGemm, 128, 0, stream>>>(x16, wk16, bk, (void*)k16, 1.0f);
    gemm512_kernel<true><<<gGemm, 128, 0, stream>>>(x16, wv16, bv, (void*)v16, 1.0f);

    // attention: B*W*H heads, 2 row-halves each
    attn_kernel<<<8 * 64 * 8 * 2, 256, 0, stream>>>(q16, k16, v16, mask, pos_bias, z16);

    // output projection (f32 out)
    gemm512_kernel<false><<<gGemm, 128, 0, stream>>>(z16, wp16, bp, (void*)out, 1.0f);
}
